// ClusterMemory_37349035606124
// MI455X (gfx1250) — matrix operand order repaired and checked by reading, not compiled, not run on hardware
//
#include <hip/hip_runtime.h>
#include <hip/hip_bf16.h>

typedef __attribute__((ext_vector_type(16))) _Float16 v16h;
typedef __attribute__((ext_vector_type(8)))  _Float16 v8h;
typedef __attribute__((ext_vector_type(8)))  float    v8f;
typedef __attribute__((ext_vector_type(4)))  float    v4f;

#define NF    2048          // feature dim (K)
#define NS    65536         // memory-bank rows (N)
#define NB    256           // batch (M)
#define NBLK  64            // N columns per block
#define KSTEP 32            // K per WMMA
#define NGEMMBLK (NS / NBLK)      // 1024 blocks
#define NPART    (NGEMMBLK * 2)   // 2 n-groups per block -> 2048 partials per row
#define INV_TEMP 20.0f

// ---------------------------------------------------------------------------
// Kernel 1: L2-normalize each input row, emit f16 copy in plain K order (no
// swizzle). The GEMM uses contiguous k = 16h + s for A and B in each 32-wide K step,
// so in the GEMM a lane's entire
// 16-element A fragment (lane<16: K 0-15; lane>=16: K 16-31)
// is ONE contiguous 32B load with zero repack VALU.
// ---------------------------------------------------------------------------
__global__ __launch_bounds__(256)
void normalize_f16_kernel(const float* __restrict__ x, _Float16* __restrict__ xh) {
    const int row = blockIdx.x;          // 256 rows
    const int tid = threadIdx.x;         // 256 threads, 8 elems each
    const float* xr = x + (size_t)row * NF;

    float vals[8];
    float ss = 0.f;
#pragma unroll
    for (int i = 0; i < 8; ++i) {
        float v = xr[tid + 256 * i];
        vals[i] = v;
        ss += v * v;
    }
    // wave32 reduce
#pragma unroll
    for (int off = 1; off < 32; off <<= 1) ss += __shfl_xor(ss, off, 32);

    __shared__ float red[8];
    __shared__ float invn;
    const int wv = tid >> 5, ln = tid & 31;
    if (ln == 0) red[wv] = ss;
    __syncthreads();
    if (tid == 0) {
        float tot = 0.f;
#pragma unroll
        for (int i = 0; i < 8; ++i) tot += red[i];
        invn = rsqrtf(tot);
    }
    __syncthreads();
    const float inv = invn;

    // store: identity mapping, each 32-step keeps its original K order
    _Float16* xo = xh + (size_t)row * NF;
    const int k32 = tid & 31;
    const int g   = k32 >> 3;                      // 0..3
    const int gg  = g;                             // 0,1,2,3 (identity)
    const int swz = (gg << 3) | (k32 & 7);
#pragma unroll
    for (int i = 0; i < 8; ++i) {
        const int k = tid + 256 * i;
        xo[(k & ~31) | swz] = (_Float16)(vals[i] * inv);
    }
}

// ---------------------------------------------------------------------------
// GEMM helpers (always inlined; ping-pong buffers for SW pipelining)
// ---------------------------------------------------------------------------
__device__ __forceinline__ void load_a(const _Float16* const aptr[4], int k0,
                                       v16h a[4]) {
#pragma unroll
    for (int mt = 0; mt < 4; ++mt)
        a[mt] = *(const v16h*)(aptr[mt] + k0);     // 32B contiguous fragment
}

__device__ __forceinline__ void load_b(const float* const bptr[2], int k0,
                                       v4f br[2][4]) {
#pragma unroll
    for (int nt = 0; nt < 2; ++nt) {
        const float* p = bptr[nt] + k0;
#pragma unroll
        for (int j = 0; j < 4; ++j) br[nt][j] = *(const v4f*)(p + 4 * j);
    }
}

__device__ __forceinline__ void do_wmma(const v16h a[4], const v4f br[2][4],
                                        v8f acc[4][2]) {
    v16h b[2];
#pragma unroll
    for (int nt = 0; nt < 2; ++nt)
#pragma unroll
        for (int i = 0; i < 4; ++i) {
            b[nt][i]      = (_Float16)br[nt][0][i];
            b[nt][i + 4]  = (_Float16)br[nt][1][i];
            b[nt][i + 8]  = (_Float16)br[nt][2][i];
            b[nt][i + 12] = (_Float16)br[nt][3][i];
        }
#pragma unroll
    for (int mt = 0; mt < 4; ++mt)
#pragma unroll
        for (int nt = 0; nt < 2; ++nt)
            acc[mt][nt] = __builtin_amdgcn_wmma_f32_16x16x32_f16(
                false, a[mt], false, b[nt], (short)0, acc[mt][nt],
                false, false);
}

// ---------------------------------------------------------------------------
// Kernel 2: fused GEMM (WMMA f16 -> f32) + per-block online-softmax partials.
// Block = 64 feature columns x ALL 256 batch rows  => features read ONCE.
// 8 waves = 4 M-groups (64 rows) x 2 N-groups (32 cols); wave does 4x2 tiles.
// K loop software-pipelined (double-buffered fragments).
// ---------------------------------------------------------------------------
__global__ __launch_bounds__(256)
void gemm_softmax_partial_kernel(const _Float16* __restrict__ xh,
                                 const float*    __restrict__ feats,
                                 const int*      __restrict__ targets,
                                 float* __restrict__ part_m,
                                 float* __restrict__ part_s,
                                 float* __restrict__ tgt_logit) {
    const int nb     = blockIdx.x;               // 0..1023
    const int n_base = nb * NBLK;
    const int tid    = threadIdx.x;
    const int wave   = tid >> 5;
    const int lane   = tid & 31;
    const int l15    = lane & 15;
    const int lhi    = lane >> 4;                // 0 or 1
    const int mg     = wave >> 1;                // 0..3  -> rows mg*64 .. +63
    const int ng     = wave & 1;                 // 0..1  -> cols n_base + ng*32 ..
    const int m_base = mg * 64;
    const int c_base = n_base + ng * 32;

    // accumulators: 4 M-tiles x 2 N-tiles, f32 16x16 each
    v8f acc[4][2];
#pragma unroll
    for (int mt = 0; mt < 4; ++mt)
#pragma unroll
        for (int nt = 0; nt < 2; ++nt)
#pragma unroll
            for (int e = 0; e < 8; ++e) acc[mt][nt][e] = 0.f;

    // A pointers into the plain-order f16 x copy (one 32B load per fragment)
    const _Float16* aptr[4];
#pragma unroll
    for (int mt = 0; mt < 4; ++mt)
        aptr[mt] = xh + (size_t)(m_base + 16 * mt + l15) * NF + lhi * 16;

    // B pointers (32x16 B layout): lane<16 -> K 0..15, lane>=16 -> K 16..31,
    // column = c_base + nt*16 + l15 (= feature row, fp32 HBM stream)
    const float* bptr[2];
#pragma unroll
    for (int nt = 0; nt < 2; ++nt)
        bptr[nt] = feats + (size_t)(c_base + 16 * nt + l15) * NF + lhi * 16;

    // -------- software-pipelined K loop (double-step, ping-pong buffers) ----
    v16h a0[4], a1[4];
    v4f  b0[2][4], b1[2][4];
    load_a(aptr, 0, a0);
    load_b(bptr, 0, b0);

    int k0 = 0;
    for (; k0 + 2 * KSTEP < NF; k0 += 2 * KSTEP) {
        load_a(aptr, k0 + KSTEP, a1);            // stage n+1 loads first
        load_b(bptr, k0 + KSTEP, b1);
        // unconditional speculative prefetch 8 steps ahead (dropped past end)
#pragma unroll
        for (int nt = 0; nt < 2; ++nt) {
            __builtin_prefetch(bptr[nt] + k0 + 8 * KSTEP, 0, 0);
            __builtin_prefetch(bptr[nt] + k0 + 9 * KSTEP, 0, 0);
        }
        do_wmma(a0, b0, acc);                    // compute step n
        load_a(aptr, k0 + 2 * KSTEP, a0);        // stage n+2 loads
        load_b(bptr, k0 + 2 * KSTEP, b0);
        do_wmma(a1, b1, acc);                    // compute step n+1
    }
    // tail: k0 == NF - 64; buf0 already holds step k0
    load_a(aptr, k0 + KSTEP, a1);
    load_b(bptr, k0 + KSTEP, b1);
    do_wmma(a0, b0, acc);
    do_wmma(a1, b1, acc);

    // --- fused per-row online-softmax partial over this wave's 32 columns ---
    // C/D layout: row = m_tile + v + 8*lhi, col = c_base + nt*16 + l15
#pragma unroll
    for (int mt = 0; mt < 4; ++mt) {
#pragma unroll
        for (int v = 0; v < 8; ++v) {
            const int r = m_base + 16 * mt + v + 8 * lhi;
            float z[2];
            float m = -3.4e38f;
#pragma unroll
            for (int nt = 0; nt < 2; ++nt) {
                z[nt] = acc[mt][nt][v] * INV_TEMP;
                m = fmaxf(m, z[nt]);
            }
            // target logit: exactly one block/lane owns each target column
            const int trel = targets[r] - c_base;
#pragma unroll
            for (int nt = 0; nt < 2; ++nt)
                if (trel == nt * 16 + l15) tgt_logit[r] = z[nt];

            // reduce over the 16 lanes of this half (cols): max then sum-exp
#pragma unroll
            for (int off = 1; off < 16; off <<= 1)
                m = fmaxf(m, __shfl_xor(m, off, 32));
            float s = 0.f;
#pragma unroll
            for (int nt = 0; nt < 2; ++nt) s += __expf(z[nt] - m);
#pragma unroll
            for (int off = 1; off < 16; off <<= 1)
                s += __shfl_xor(s, off, 32);

            if (l15 == 0) {
                const int pidx = nb * 2 + ng;
                part_m[(size_t)r * NPART + pidx] = m;
                part_s[(size_t)r * NPART + pidx] = s;
            }
        }
    }
}

// ---------------------------------------------------------------------------
// Kernel 3: combine 2048 (m,s) partials per row -> NLL per row
// ---------------------------------------------------------------------------
__global__ __launch_bounds__(256)
void row_reduce_kernel(const float* __restrict__ part_m,
                       const float* __restrict__ part_s,
                       const float* __restrict__ tgt_logit,
                       float* __restrict__ nll) {
    const int r   = blockIdx.x;
    const int tid = threadIdx.x;
    const float* pm = part_m + (size_t)r * NPART;
    const float* ps = part_s + (size_t)r * NPART;

    float lm[8], ls[8];
    float m = -3.4e38f;
#pragma unroll
    for (int i = 0; i < 8; ++i) {
        lm[i] = pm[tid + 256 * i];
        ls[i] = ps[tid + 256 * i];
        m = fmaxf(m, lm[i]);
    }
#pragma unroll
    for (int off = 1; off < 32; off <<= 1) m = fmaxf(m, __shfl_xor(m, off, 32));

    __shared__ float sbuf[8];
    __shared__ float Msh;
    const int wv = tid >> 5, ln = tid & 31;
    if (ln == 0) sbuf[wv] = m;
    __syncthreads();
    if (tid == 0) {
        float mm = sbuf[0];
#pragma unroll
        for (int i = 1; i < 8; ++i) mm = fmaxf(mm, sbuf[i]);
        Msh = mm;
    }
    __syncthreads();
    const float M = Msh;

    float s = 0.f;
#pragma unroll
    for (int i = 0; i < 8; ++i) s += ls[i] * __expf(lm[i] - M);
#pragma unroll
    for (int off = 1; off < 32; off <<= 1) s += __shfl_xor(s, off, 32);
    if (ln == 0) sbuf[wv] = s;
    __syncthreads();
    if (tid == 0) {
        float S = 0.f;
#pragma unroll
        for (int i = 0; i < 8; ++i) S += sbuf[i];
        nll[r] = -(tgt_logit[r] - M - __logf(S));
    }
}

// ---------------------------------------------------------------------------
// Kernel 4: mean over 256 NLLs -> scalar loss
// ---------------------------------------------------------------------------
__global__ __launch_bounds__(256)
void mean_kernel(const float* __restrict__ nll, float* __restrict__ out) {
    const int tid = threadIdx.x;
    float v = nll[tid];
#pragma unroll
    for (int off = 1; off < 32; off <<= 1) v += __shfl_xor(v, off, 32);
    __shared__ float sbuf[8];
    if ((tid & 31) == 0) sbuf[tid >> 5] = v;
    __syncthreads();
    if (tid == 0) {
        float t = 0.f;
#pragma unroll
        for (int i = 0; i < 8; ++i) t += sbuf[i];
        out[0] = t * (1.0f / NB);
    }
}

extern "C" void kernel_launch(void* const* d_in, const int* in_sizes, int n_in,
                              void* d_out, int out_size, void* d_ws, size_t ws_size,
                              hipStream_t stream) {
    (void)in_sizes; (void)n_in; (void)out_size; (void)ws_size;
    const float* inputs  = (const float*)d_in[0];   // [256, 2048] f32
    const int*   targets = (const int*)d_in[1];     // [256] i32
    // d_in[2] = cameras (unused by reference loss)
    const float* feats   = (const float*)d_in[3];   // [65536, 2048] f32

    char* ws = (char*)d_ws;
    _Float16* xh        = (_Float16*)(ws);                       // 1 MB (plain K order)
    float*    part_m    = (float*)(ws + (1u << 20));             // 2 MB
    float*    part_s    = (float*)(ws + (3u << 20));             // 2 MB
    float*    tgt_logit = (float*)(ws + (5u << 20));             // 1 KB
    float*    nll       = (float*)(ws + (5u << 20) + 4096);      // 1 KB

    normalize_f16_kernel<<<NB, 256, 0, stream>>>(inputs, xh);
    gemm_softmax_partial_kernel<<<NGEMMBLK, 256, 0, stream>>>(
        xh, feats, targets, part_m, part_s, tgt_logit);
    row_reduce_kernel<<<NB, 256, 0, stream>>>(part_m, part_s, tgt_logit, nll);
    mean_kernel<<<1, 256, 0, stream>>>(nll, (float*)d_out);
}
